// DeepseekMoEwithCache_29429115912763
// MI455X (gfx1250) — matrix operand order repaired and checked by reading, not compiled, not run on hardware
//
#include <hip/hip_runtime.h>
#include <math.h>

// ---------------- problem constants (DeepSeek MoE reference) ----------------
#define E_EXPERTS 8
#define HDIM      2048
#define IDIM      1408
#define ISDIM     2816      // 2 * IDIM (shared experts)
#define TTOK      1024      // B*S = 2*512

#define AROWB  80           // LDS row stride (bytes) for staged A tile: 16B-aligned,
                            // 20 dwords -> conflict-free b128 reads across 16 lanes
#define ATILEB (32 * AROWB) // one staged 32x32 bf16 tile

typedef __attribute__((ext_vector_type(16))) __bf16 v16bf;
typedef __attribute__((ext_vector_type(8)))  float  v8f;
typedef __attribute__((ext_vector_type(4)))  float  v4f;   // native vec for NT builtins

// ---- pack two fp32 -> one dword of two bf16 (round-to-nearest + v_perm) ----
__device__ __forceinline__ unsigned pk_bf16(float a, float b) {
#if __has_builtin(__builtin_amdgcn_cvt_pk_bf16_f32)
    auto t = __builtin_amdgcn_cvt_pk_bf16_f32(a, b);
    unsigned u; __builtin_memcpy(&u, &t, 4); return u;
#else
    union { float f; unsigned u; } x, y; x.f = a; y.f = b;
    return __builtin_amdgcn_perm(y.u + 0x8000u, x.u + 0x8000u, 0x07060302u);
#endif
}

__device__ __forceinline__ __bf16 f2bf(float f) {
    union { float f; unsigned u; } a; a.f = f;
    unsigned r = a.u + 0x7FFFu + ((a.u >> 16) & 1u);
    unsigned short h = (unsigned short)(r >> 16);
    union { unsigned short s; __bf16 b; } o; o.s = h; return o.b;
}

__device__ __forceinline__ float fast_rcp(float x) {
#if __has_builtin(__builtin_amdgcn_rcpf)
    return __builtin_amdgcn_rcpf(x);
#else
    return 1.0f / x;
#endif
}

__device__ __forceinline__ v8f wmma_bf16(v16bf a, v16bf b, v8f c) {
    return __builtin_amdgcn_wmma_f32_16x16x32_bf16(
        false, a, false, b, (short)0, c, false, false);
}

// B fragment from bf16 K-major [K][ldn]: lane L -> n = n0+L%16, elems j -> k = k0 + 16*(j/8) + 8*(L/16) + j%8.
__device__ __forceinline__ v16bf load_b_kmaj(const __bf16* __restrict__ base, int ldn,
                                             int k0, int n0, int lane) {
    const __bf16* p = base + (size_t)(k0 + 8 * (lane >> 4)) * ldn + n0 + (lane & 15);
    union { v16bf v; __bf16 e[16]; } r;
#pragma unroll
    for (int j = 0; j < 16; ++j) r.e[j] = p[(size_t)((j & 7) + 16 * (j >> 3)) * ldn];
    return r.v;
}

// ---- cooperative A-tile staging: 256 threads convert a 32(M)x32(K) fp32 tile ----
__device__ __forceinline__ v4f a_stage_load(const float* __restrict__ src, int ld,
                                            int row0, int k0, int tid) {
    int r = tid >> 3, c = tid & 7;              // 32 rows x (8 threads * 4 floats)
    // streamed weights: keep them out of the near cache (reused only via L2)
    return __builtin_nontemporal_load(
        (const v4f*)(src + (size_t)(row0 + r) * ld + k0 + c * 4));
}
__device__ __forceinline__ void a_stage_prefetch(const float* __restrict__ src, int ld,
                                                 int row0, int k0, int tid) {
    int r = tid >> 3, c = tid & 7;
    __builtin_prefetch(src + (size_t)(row0 + r) * ld + k0 + c * 4, 0, 1);
}
__device__ __forceinline__ void a_stage_store(unsigned char* buf, v4f f, int tid) {
    int r = tid >> 3, c = tid & 7;
    uint2 d; d.x = pk_bf16(f.x, f.y); d.y = pk_bf16(f.z, f.w);
    *(uint2*)(buf + r * AROWB + c * 8) = d;
}
// A fragment from staged LDS tile (row m holds k0..k0+31 as bf16 in 64 of 80 bytes)
// ISA A layout: lane L -> m = L%16; elems j: k = 16*(j/8) + 8*(L/16) + j%8.
__device__ __forceinline__ v16bf a_frag_lds(const unsigned char* buf, int mrow, int lane) {
    int m   = mrow + (lane & 15);
    int off = m * AROWB + (lane >> 4) * 16;
    union { v16bf v; uint4 q[2]; } r;
    r.q[0] = *(const uint4*)(buf + off);
    r.q[1] = *(const uint4*)(buf + off + 32);
    return r.v;
}

// ---------------------------------------------------------------------------
// Kernel 1: router logits + softmax + top2 + normalized combine weights
// ---------------------------------------------------------------------------
__global__ __launch_bounds__(256)
void moe_router_kernel(const float* __restrict__ x, const float* __restrict__ rw,
                       float* __restrict__ cw, float* __restrict__ logits_out) {
    int t    = blockIdx.x;
    int wave = threadIdx.x >> 5;   // expert index (8 waves = 8 experts)
    int lane = threadIdx.x & 31;
    const float* xr = x + (size_t)t * HDIM;
    const float* wr = rw + (size_t)wave * HDIM;
    float s = 0.f;
    for (int h = lane; h < HDIM; h += 32) s += xr[h] * wr[h];
#pragma unroll
    for (int o = 16; o >= 1; o >>= 1) s += __shfl_xor(s, o, 32);

    __shared__ float lg[E_EXPERTS];
    if (lane == 0) {
        lg[wave] = s;
        logits_out[(size_t)t * E_EXPERTS + wave] = s;
    }
    __syncthreads();

    if (threadIdx.x == 0) {
        float m = lg[0];
#pragma unroll
        for (int e = 1; e < E_EXPERTS; ++e) m = fmaxf(m, lg[e]);
        float p[E_EXPERTS]; float sum = 0.f;
#pragma unroll
        for (int e = 0; e < E_EXPERTS; ++e) { p[e] = expf(lg[e] - m); sum += p[e]; }
        float inv = fast_rcp(sum);
#pragma unroll
        for (int e = 0; e < E_EXPERTS; ++e) p[e] *= inv;
        int i1 = 0; float v1 = p[0];
#pragma unroll
        for (int e = 1; e < E_EXPERTS; ++e) if (p[e] > v1) { v1 = p[e]; i1 = e; }
        int i2 = -1; float v2 = -1.f;
#pragma unroll
        for (int e = 0; e < E_EXPERTS; ++e) if (e != i1 && p[e] > v2) { v2 = p[e]; i2 = e; }
        float norm = fast_rcp(v1 + v2 + 1e-6f);
#pragma unroll
        for (int e = 0; e < E_EXPERTS; ++e) cw[(size_t)t * E_EXPERTS + e] = 0.f;
        cw[(size_t)t * E_EXPERTS + i1] = v1 * norm;
        cw[(size_t)t * E_EXPERTS + i2] = v2 * norm;
    }
}

// ---------------------------------------------------------------------------
// Kernel 2: x [T][H] fp32 -> xT [H][T] bf16 (LDS 32x33 tile transpose)
// ---------------------------------------------------------------------------
__global__ __launch_bounds__(256)
void transpose_x_kernel(const float* __restrict__ x, __bf16* __restrict__ xT) {
    __shared__ float tile[32][33];
    int t0 = blockIdx.x * 32, h0 = blockIdx.y * 32;
    int tr = threadIdx.x & 31;
    int rr = threadIdx.x >> 5;
#pragma unroll
    for (int i = 0; i < 4; ++i) {
        int tl = rr + i * 8;
        tile[tl][tr] = x[(size_t)(t0 + tl) * HDIM + h0 + tr];
    }
    __syncthreads();
#pragma unroll
    for (int i = 0; i < 4; ++i) {
        int hl = rr + i * 8;
        xT[(size_t)(h0 + hl) * TTOK + t0 + tr] = f2bf(tile[tr][hl]);
    }
}

// ---------------------------------------------------------------------------
// Kernel 3: transposed fused gate+up GEMM + SiLU + cw scale -> bf16 K-major
//   dstT[e][i][t] = silu(Wg[e] x^T) * (Wu[e] x^T) * cw[t][e]
// M = weight rows, N = tokens, K = H. 8 waves along N; A staged via LDS (2 bufs).
// ---------------------------------------------------------------------------
__global__ __launch_bounds__(256)
void moe_gateup_t_kernel(const __bf16* __restrict__ xT,       // [H][T] bf16
                         const float* __restrict__ wg_all,
                         const float* __restrict__ wu_all,
                         const float* __restrict__ cw,        // null -> 1.0
                         __bf16* __restrict__ dstT_all,       // [e][ncols][T]
                         int ncols) {
    __shared__ __align__(16) unsigned char sa[2][2][ATILEB];
    int e = blockIdx.z;
    const float* wg = wg_all + (size_t)e * ncols * HDIM;
    const float* wu = wu_all + (size_t)e * ncols * HDIM;
    __bf16* dstT = dstT_all + (size_t)e * ncols * TTOK;

    int tid = threadIdx.x, wave = tid >> 5, lane = tid & 31;
    int im = blockIdx.y * 32;                 // weight-row tile (M)
    int tn = blockIdx.x * 256 + wave * 32;    // token tile (N)

    v8f accg[2][2] = {};
    v8f accu[2][2] = {};

    a_stage_store(sa[0][0], a_stage_load(wg, HDIM, im, 0, tid), tid);
    a_stage_store(sa[0][1], a_stage_load(wu, HDIM, im, 0, tid), tid);
    int buf = 0;
    for (int k0 = 0; k0 < HDIM; k0 += 32) {
        __syncthreads();
        bool more = (k0 + 32) < HDIM;
        v4f ng, nu;
        if (more) {
            ng = a_stage_load(wg, HDIM, im, k0 + 32, tid);
            nu = a_stage_load(wu, HDIM, im, k0 + 32, tid);
        }
        if (k0 + 64 < HDIM) {
            a_stage_prefetch(wg, HDIM, im, k0 + 64, tid);
            a_stage_prefetch(wu, HDIM, im, k0 + 64, tid);
        }
        v16bf b0  = load_b_kmaj(xT, TTOK, k0, tn,      lane);
        v16bf b1  = load_b_kmaj(xT, TTOK, k0, tn + 16, lane);
        v16bf ag0 = a_frag_lds(sa[buf][0], 0,  lane);
        v16bf ag1 = a_frag_lds(sa[buf][0], 16, lane);
        v16bf au0 = a_frag_lds(sa[buf][1], 0,  lane);
        v16bf au1 = a_frag_lds(sa[buf][1], 16, lane);
        accg[0][0] = wmma_bf16(ag0, b0, accg[0][0]);
        accg[0][1] = wmma_bf16(ag0, b1, accg[0][1]);
        accg[1][0] = wmma_bf16(ag1, b0, accg[1][0]);
        accg[1][1] = wmma_bf16(ag1, b1, accg[1][1]);
        accu[0][0] = wmma_bf16(au0, b0, accu[0][0]);
        accu[0][1] = wmma_bf16(au0, b1, accu[0][1]);
        accu[1][0] = wmma_bf16(au1, b0, accu[1][0]);
        accu[1][1] = wmma_bf16(au1, b1, accu[1][1]);
        if (more) {
            a_stage_store(sa[buf ^ 1][0], ng, tid);
            a_stage_store(sa[buf ^ 1][1], nu, tid);
        }
        buf ^= 1;
    }

    // C layout: VGPR r, lane L -> m = base + r + 8*(L/16), n = base + L%16
    int half = lane >> 4, nl = lane & 15;
    float cwv[2];
    cwv[0] = cw ? cw[(size_t)(tn + nl)      * E_EXPERTS + e] : 1.0f;
    cwv[1] = cw ? cw[(size_t)(tn + 16 + nl) * E_EXPERTS + e] : 1.0f;
#pragma unroll
    for (int rt = 0; rt < 2; ++rt) {
#pragma unroll
        for (int ct = 0; ct < 2; ++ct) {
            v8f g = accg[rt][ct];
            v8f u = accu[rt][ct];
#pragma unroll
            for (int r = 0; r < 8; ++r) {
                int m = im + rt * 16 + r + 8 * half;
                int n = tn + ct * 16 + nl;
                float gv = g[r];
                float hv = gv * fast_rcp(1.f + __expf(-gv)) * u[r] * cwv[ct];
                __builtin_nontemporal_store(f2bf(hv), dstT + (size_t)m * TTOK + n);
            }
        }
    }
}

// ---------------------------------------------------------------------------
// Kernel 4: transposed down projection over routed + shared experts
// ---------------------------------------------------------------------------
__device__ __forceinline__ void down_segment(const float* __restrict__ A, int lda,
                                             const __bf16* __restrict__ Bt, int ksteps,
                                             int dm, int tn, int tid, int lane,
                                             unsigned char (*sa)[ATILEB], int& buf,
                                             v8f acc[2][2]) {
    // prefill writes the buffer whose last reads were guarded by the previous
    // iteration's barrier (same invariant as the steady-state loop).
    a_stage_store(sa[buf], a_stage_load(A, lda, dm, 0, tid), tid);
    for (int s = 0; s < ksteps; ++s) {
        __syncthreads();
        int k0 = s * 32;
        bool more = (s + 1) < ksteps;
        v4f nf;
        if (more) nf = a_stage_load(A, lda, dm, k0 + 32, tid);
        if (s + 2 < ksteps) a_stage_prefetch(A, lda, dm, k0 + 64, tid);
        v16bf b0 = load_b_kmaj(Bt, TTOK, k0, tn,      lane);
        v16bf b1 = load_b_kmaj(Bt, TTOK, k0, tn + 16, lane);
        v16bf a0 = a_frag_lds(sa[buf], 0,  lane);
        v16bf a1 = a_frag_lds(sa[buf], 16, lane);
        acc[0][0] = wmma_bf16(a0, b0, acc[0][0]);
        acc[0][1] = wmma_bf16(a0, b1, acc[0][1]);
        acc[1][0] = wmma_bf16(a1, b0, acc[1][0]);
        acc[1][1] = wmma_bf16(a1, b1, acc[1][1]);
        if (more) a_stage_store(sa[buf ^ 1], nf, tid);
        buf ^= 1;
    }
}

__global__ __launch_bounds__(256)
void moe_down_t_kernel(const __bf16* __restrict__ hcatT,   // [E][I][T]
                       const __bf16* __restrict__ shT,     // [IS][T]
                       const float* __restrict__ w_down,   // [E][H][I]
                       const float* __restrict__ sw_down,  // [H][IS]
                       float* __restrict__ out) {          // [T][H]
    __shared__ __align__(16) unsigned char smem[256 * 33 * 4]; // staging + epilogue
    unsigned char (*sa)[ATILEB] = (unsigned char (*)[ATILEB])smem;

    int tid = threadIdx.x, wave = tid >> 5, lane = tid & 31;
    int dm = blockIdx.y * 32;                 // hidden tile (M)
    int tn = blockIdx.x * 256 + wave * 32;    // token tile (N)

    v8f acc[2][2] = {};
    int buf = 0;
    for (int e = 0; e < E_EXPERTS; ++e)
        down_segment(w_down + (size_t)e * HDIM * IDIM, IDIM,
                     hcatT + (size_t)e * IDIM * TTOK, IDIM / 32,
                     dm, tn, tid, lane, sa, buf, acc);
    down_segment(sw_down, ISDIM, shT, ISDIM / 32, dm, tn, tid, lane, sa, buf, acc);

    // stage C (32 d x 256 t) into LDS, then write coalesced rows of out[t][d]
    __syncthreads();
    float* tile = (float*)smem;
    int half = lane >> 4, nl = lane & 15;
#pragma unroll
    for (int rt = 0; rt < 2; ++rt) {
#pragma unroll
        for (int ct = 0; ct < 2; ++ct) {
            v8f c = acc[rt][ct];
#pragma unroll
            for (int r = 0; r < 8; ++r) {
                int dl = rt * 16 + r + 8 * half;        // 0..31
                int tl = wave * 32 + ct * 16 + nl;      // 0..255
                tile[tl * 33 + dl] = c[r];
            }
        }
    }
    __syncthreads();
    int t = blockIdx.x * 256 + tid;
    float* orow = out + (size_t)t * HDIM + dm;
#pragma unroll
    for (int v = 0; v < 8; ++v) {
        v4f f;
        f.x = tile[tid * 33 + 4 * v + 0];
        f.y = tile[tid * 33 + 4 * v + 1];
        f.z = tile[tid * 33 + 4 * v + 2];
        f.w = tile[tid * 33 + 4 * v + 3];
        __builtin_nontemporal_store(f, (v4f*)(orow + 4 * v));
    }
}

// ---------------------------------------------------------------------------
extern "C" void kernel_launch(void* const* d_in, const int* in_sizes, int n_in,
                              void* d_out, int out_size, void* d_ws, size_t ws_size,
                              hipStream_t stream) {
    const float* x        = (const float*)d_in[0];  // [T,H]
    const float* router_w = (const float*)d_in[1];  // [E,H]
    const float* w_gate   = (const float*)d_in[2];  // [E,I,H]
    const float* w_up     = (const float*)d_in[3];  // [E,I,H]
    const float* w_down   = (const float*)d_in[4];  // [E,H,I]
    const float* sw_gate  = (const float*)d_in[5];  // [IS,H]
    const float* sw_up    = (const float*)d_in[6];  // [IS,H]
    const float* sw_down  = (const float*)d_in[7];  // [H,IS]

    float* out    = (float*)d_out;                  // final [T*H]
    float* logits = out + (size_t)TTOK * HDIM;      // router_logits [T*E]

    // workspace (~33 MB): cw | xT bf16 | hcatT bf16 | shT bf16
    char* ws = (char*)d_ws;
    float*  cw    = (float*)ws;                                       // T*E
    __bf16* xT    = (__bf16*)(ws + (size_t)TTOK * E_EXPERTS * 4);     // [H][T]
    __bf16* hcatT = xT + (size_t)HDIM * TTOK;                         // [E][I][T]
    __bf16* shT   = hcatT + (size_t)E_EXPERTS * IDIM * TTOK;          // [IS][T]

    moe_router_kernel<<<TTOK, 256, 0, stream>>>(x, router_w, cw, logits);

    dim3 gT(TTOK / 32, HDIM / 32);
    transpose_x_kernel<<<gT, 256, 0, stream>>>(x, xT);

    dim3 gA(TTOK / 256, IDIM / 32, E_EXPERTS);
    moe_gateup_t_kernel<<<gA, 256, 0, stream>>>(xT, w_gate, w_up, cw, hcatT, IDIM);

    dim3 gS(TTOK / 256, ISDIM / 32, 1);
    moe_gateup_t_kernel<<<gS, 256, 0, stream>>>(xT, sw_gate, sw_up, nullptr, shT, ISDIM);

    dim3 gB(TTOK / 256, HDIM / 32, 1);
    moe_down_t_kernel<<<gB, 256, 0, stream>>>(hcatT, shT, w_down, sw_down, out);
}
